// GnBlock_11373073400277
// MI455X (gfx1250) — matrix operand order repaired and checked by reading, not compiled, not run on hardware
//
#include <hip/hip_runtime.h>
#include <hip/hip_bf16.h>

// ---------------------------------------------------------------------------
// GraphNet block for MI455X (gfx1250, wave32).
// Edge MLP + scatter-add + Node MLP, all GEMMs via v_wmma_f32_16x16x32_bf16.
// ---------------------------------------------------------------------------

typedef __attribute__((ext_vector_type(16))) __bf16 v16bf;
typedef __attribute__((ext_vector_type(8)))  float  v8f;

#define H          128
#define N_NODES    20000
#define N_EDGES    200000
#define LN_EPS     1e-5f

// bf16 weight-fragment offsets (units: halves). One k-tile = 8 ntiles*32 lanes*16 = 4096 halves.
#define KT_E1 12
#define KT_EX 4
#define KT_N1 8
#define FOFF_E1 0
#define FOFF_E2 (12*4096)
#define FOFF_E3 (16*4096)
#define FOFF_E4 (20*4096)
#define FOFF_N1 (24*4096)
#define FOFF_N2 (32*4096)
#define FOFF_N3 (36*4096)
#define FOFF_N4 (40*4096)
#define WFRAG_HALVES (44*4096)

union ABFrag { v16bf v; uint4 q[2]; };

__device__ __forceinline__ unsigned short f2bf(float f) {
    union { float f; unsigned u; } c; c.f = f;
    unsigned u = c.u + 0x7FFFu + ((c.u >> 16) & 1u);   // round-to-nearest-even
    return (unsigned short)(u >> 16);
}

__device__ __forceinline__ void wave_sync() {
    __builtin_amdgcn_wave_barrier();
    asm volatile("s_wait_dscnt 0" ::: "memory");
    __builtin_amdgcn_wave_barrier();
}

__device__ __forceinline__ v8f wmma_bf16(v16bf a, v16bf b, v8f c) {
    return __builtin_amdgcn_wmma_f32_16x16x32_bf16(
        /*neg_a=*/false, a, /*neg_b=*/false, b,
        /*c_mod=*/(short)0, c, /*reuse_a=*/false, /*reuse_b=*/false);
}

// ---------------------------------------------------------------------------
// Per-wave LDS slab (no cross-wave sharing -> no block barriers needed).
// bufA doubles as the f32 LayerNorm staging buffer (16x128 f32 = 8KB <= 12KB).
// ---------------------------------------------------------------------------
struct __align__(16) WaveLDS {
    unsigned short bufA[16 * 384];   // 12 KB : layer-1 input / even-layer acts / f32 LN stage
    unsigned short bufH[16 * 128];   //  4 KB : odd-layer activations
    float mu[16];
    float rs[16];
    int   eidx[32];                  // e0[0..15], e1[0..15] (edge kernel only)
};

// ---------------------------------------------------------------------------
// Wave-level GEMM: 16 rows x 128 cols, K = 32*ktiles.
// A: bf16 row-major in LDS (stride strideH halves). B: pre-swizzled bf16 frags.
//   A-frag (16x32): lane(0..15)=row m, halves 0..7 -> K 0..7, 8..15 -> K 16..23;
//                   lanes 16..31 same rows, K shifted by +8.  => 2x ds_load_b128.
//   B-frag (32x16): lane -> col n=lane&15, halves 0..7 -> K 0..7, 8..15 -> K 16..23 (+8 hi lanes),
//                   stored contiguous per lane               => 2x global_load_b128.
// ---------------------------------------------------------------------------
__device__ __forceinline__ void gemm_layer(const unsigned short* ldsIn, int strideH,
                                           const unsigned short* wfrag, int ktiles,
                                           int lane, v8f acc[8]) {
    const int m  = lane & 15;
    const int hi = lane >> 4;
    #pragma unroll
    for (int nt = 0; nt < 8; ++nt)
        acc[nt] = v8f{0.f,0.f,0.f,0.f,0.f,0.f,0.f,0.f};

    for (int kt = 0; kt < ktiles; ++kt) {
        const unsigned short* pa = ldsIn + m * strideH + kt * 32 + hi * 8;
        ABFrag a;
        a.q[0] = *(const uint4*)(pa);        // K 0..7   (+8 for hi lanes)
        a.q[1] = *(const uint4*)(pa + 16);   // K 16..23 (+8 for hi lanes)
        const unsigned short* pb = wfrag + kt * 4096 + lane * 16;
        #pragma unroll
        for (int nt = 0; nt < 8; ++nt) {
            ABFrag b;
            b.q[0] = *(const uint4*)(pb + nt * 512);
            b.q[1] = *(const uint4*)(pb + nt * 512 + 8);
            acc[nt] = wmma_bf16(a.v, b.v, acc[nt]);
        }
    }
}

// bias + optional ReLU, write bf16 activations for the next layer.
// C/D layout: vgpr r, lane -> n = 16*nt + (lane&15), m = r + (lane<16 ? 0 : 8).
__device__ __forceinline__ void epi_bf16(v8f acc[8], const float* __restrict__ bias,
                                         bool relu, unsigned short* ldsOut, int lane) {
    const int n0 = lane & 15;
    const int hi = lane >> 4;
    #pragma unroll
    for (int nt = 0; nt < 8; ++nt) {
        const int n = nt * 16 + n0;
        const float bb = bias[n];
        #pragma unroll
        for (int r = 0; r < 8; ++r) {
            float v = acc[nt][r] + bb;
            if (relu) v = v > 0.f ? v : 0.f;
            ldsOut[(r + hi * 8) * 128 + n] = f2bf(v);
        }
    }
}

// bias only, write f32 for LayerNorm.
__device__ __forceinline__ void epi_f32(v8f acc[8], const float* __restrict__ bias,
                                        float* ldsF, int lane) {
    const int n0 = lane & 15;
    const int hi = lane >> 4;
    #pragma unroll
    for (int nt = 0; nt < 8; ++nt) {
        const int n = nt * 16 + n0;
        const float bb = bias[n];
        #pragma unroll
        for (int r = 0; r < 8; ++r)
            ldsF[(r + hi * 8) * 128 + n] = acc[nt][r] + bb;
    }
}

__device__ __forceinline__ void layernorm_stats(const float* ldsF, float* mu, float* rs, int lane) {
    if (lane < 16) {
        float s = 0.f, ss = 0.f;
        #pragma unroll 8
        for (int c = 0; c < 128; ++c) {
            float v = ldsF[lane * 128 + c];
            s += v; ss += v * v;
        }
        float m  = s * (1.f / 128.f);
        float var = ss * (1.f / 128.f) - m * m;
        mu[lane] = m;
        rs[lane] = rsqrtf(var + LN_EPS);
    }
}

// ---------------------------------------------------------------------------
// Weight prep: fp32 (K x 128) row-major -> bf16 fragment layout in workspace.
// grid (44 ktiles, 8 ntiles), block 512: thread -> (lane = t&31, i = t>>5).
// ---------------------------------------------------------------------------
__global__ void prep_weights(const float* __restrict__ ew1, const float* __restrict__ ew2,
                             const float* __restrict__ ew3, const float* __restrict__ ew4,
                             const float* __restrict__ nw1, const float* __restrict__ nw2,
                             const float* __restrict__ nw3, const float* __restrict__ nw4,
                             unsigned short* __restrict__ wfrag) {
    const int ktg  = blockIdx.x;
    const int nt   = blockIdx.y;
    const int lane = threadIdx.x & 31;
    const int i    = threadIdx.x >> 5;

    const float* Wm; int kt;
    if      (ktg < 12) { Wm = ew1; kt = ktg;      }
    else if (ktg < 16) { Wm = ew2; kt = ktg - 12; }
    else if (ktg < 20) { Wm = ew3; kt = ktg - 16; }
    else if (ktg < 24) { Wm = ew4; kt = ktg - 20; }
    else if (ktg < 32) { Wm = nw1; kt = ktg - 24; }
    else if (ktg < 36) { Wm = nw2; kt = ktg - 32; }
    else if (ktg < 40) { Wm = nw3; kt = ktg - 36; }
    else               { Wm = nw4; kt = ktg - 40; }

    const int k = kt * 32 + (i & 7) + 8 * (lane >> 4) + 16 * (i >> 3);
    const int n = nt * 16 + (lane & 15);
    wfrag[ktg * 4096 + nt * 512 + lane * 16 + i] = f2bf(Wm[k * 128 + n]);
}

__global__ void zero_f32(float* __restrict__ p, int n) {
    int i = blockIdx.x * blockDim.x + threadIdx.x;
    if (i < n) p[i] = 0.f;
}

// ---------------------------------------------------------------------------
// Edge kernel: 2 waves/block, each wave owns 16 edges.
// gather -> 4-layer MLP (WMMA bf16) -> LayerNorm -> residual store + atomic scatter.
// ---------------------------------------------------------------------------
__global__ void __launch_bounds__(64)
edge_kernel(const float* __restrict__ x_node, const float* __restrict__ x_edge,
            const long long* __restrict__ edge_index,
            const unsigned short* __restrict__ wfrag,
            const float* __restrict__ eb1, const float* __restrict__ eb2,
            const float* __restrict__ eb3, const float* __restrict__ eb4,
            const float* __restrict__ eg,  const float* __restrict__ ebeta,
            float* __restrict__ agg, float* __restrict__ out_edge) {
    __shared__ WaveLDS slab[2];
    const int wave = threadIdx.x >> 5;
    const int lane = threadIdx.x & 31;
    WaveLDS& L = slab[wave];

    const long long ebase = (long long)(blockIdx.x * 2 + wave) * 16;

    // endpoint indices: lanes 0..15 -> e0 of row=lane, lanes 16..31 -> e1.
    {
        const int hi  = lane >> 4;
        const int row = lane & 15;
        L.eidx[lane] = (int)edge_index[(ebase + row) * 2 + hi];
    }
    wave_sync();

    // Assemble A1 = [x_node[e0] | x_node[e1] | x_edge] as bf16 (16 x 384).
    for (int row = 0; row < 16; ++row) {
        const float* s0 = x_node + (long long)L.eidx[row]      * H;
        const float* s1 = x_node + (long long)L.eidx[16 + row] * H;
        const float* s2 = x_edge + (ebase + row) * H;
        #pragma unroll
        for (int it = 0; it < 4; ++it) {
            int c = it * 32 + lane;
            L.bufA[row * 384 +       c] = f2bf(s0[c]);
        }
        #pragma unroll
        for (int it = 0; it < 4; ++it) {
            int c = it * 32 + lane;
            L.bufA[row * 384 + 128 + c] = f2bf(s1[c]);
        }
        #pragma unroll
        for (int it = 0; it < 4; ++it) {
            int c = it * 32 + lane;
            L.bufA[row * 384 + 256 + c] = f2bf(s2[c]);
        }
    }
    wave_sync();

    v8f acc[8];
    gemm_layer(L.bufA, 384, wfrag + FOFF_E1, KT_E1, lane, acc);
    epi_bf16(acc, eb1, true, L.bufH, lane);  wave_sync();
    gemm_layer(L.bufH, 128, wfrag + FOFF_E2, KT_EX, lane, acc);
    epi_bf16(acc, eb2, true, L.bufA, lane);  wave_sync();
    gemm_layer(L.bufA, 128, wfrag + FOFF_E3, KT_EX, lane, acc);
    epi_bf16(acc, eb3, true, L.bufH, lane);  wave_sync();
    gemm_layer(L.bufH, 128, wfrag + FOFF_E4, KT_EX, lane, acc);

    float* F = reinterpret_cast<float*>(L.bufA);     // 16x128 f32 stage
    epi_f32(acc, eb4, F, lane);              wave_sync();

    layernorm_stats(F, L.mu, L.rs, lane);    wave_sync();

    // gamma/beta slices for this lane's 4 columns
    float gv[4], bv[4];
    #pragma unroll
    for (int it = 0; it < 4; ++it) { gv[it] = eg[it * 32 + lane]; bv[it] = ebeta[it * 32 + lane]; }

    for (int row = 0; row < 16; ++row) {
        const long long e = ebase + row;
        const float m = L.mu[row], r = L.rs[row];
        const long long i0 = (long long)L.eidx[row]      * H;
        const long long i1 = (long long)L.eidx[16 + row] * H;
        #pragma unroll
        for (int it = 0; it < 4; ++it) {
            const int c = it * 32 + lane;
            const float y = (F[row * 128 + c] - m) * r * gv[it] + bv[it];
            out_edge[e * H + c] = x_edge[e * H + c] + y;   // residual edge output
            atomicAdd(&agg[i0 + c], y);                    // segment_sum over e0
            atomicAdd(&agg[i1 + c], y);                    // segment_sum over e1
        }
    }
}

// ---------------------------------------------------------------------------
// Node kernel: 2 waves/block, each wave owns 16 nodes. Input = [x_node | agg].
// ---------------------------------------------------------------------------
__global__ void __launch_bounds__(64)
node_kernel(const float* __restrict__ x_node, const float* __restrict__ agg,
            const unsigned short* __restrict__ wfrag,
            const float* __restrict__ nb1, const float* __restrict__ nb2,
            const float* __restrict__ nb3, const float* __restrict__ nb4,
            const float* __restrict__ ng,  const float* __restrict__ nbeta,
            float* __restrict__ out_node) {
    __shared__ WaveLDS slab[2];
    const int wave = threadIdx.x >> 5;
    const int lane = threadIdx.x & 31;
    WaveLDS& L = slab[wave];

    const long long nbase = (long long)(blockIdx.x * 2 + wave) * 16;

    // Assemble A1 = [x_node | agg] as bf16 (16 x 256).
    for (int row = 0; row < 16; ++row) {
        const float* s0 = x_node + (nbase + row) * H;
        const float* s1 = agg    + (nbase + row) * H;
        #pragma unroll
        for (int it = 0; it < 4; ++it) {
            int c = it * 32 + lane;
            L.bufA[row * 256 +       c] = f2bf(s0[c]);
        }
        #pragma unroll
        for (int it = 0; it < 4; ++it) {
            int c = it * 32 + lane;
            L.bufA[row * 256 + 128 + c] = f2bf(s1[c]);
        }
    }
    wave_sync();

    v8f acc[8];
    gemm_layer(L.bufA, 256, wfrag + FOFF_N1, KT_N1, lane, acc);
    epi_bf16(acc, nb1, true, L.bufH, lane);  wave_sync();
    gemm_layer(L.bufH, 128, wfrag + FOFF_N2, KT_EX, lane, acc);
    epi_bf16(acc, nb2, true, L.bufA, lane);  wave_sync();
    gemm_layer(L.bufA, 128, wfrag + FOFF_N3, KT_EX, lane, acc);
    epi_bf16(acc, nb3, true, L.bufH, lane);  wave_sync();
    gemm_layer(L.bufH, 128, wfrag + FOFF_N4, KT_EX, lane, acc);

    float* F = reinterpret_cast<float*>(L.bufA);
    epi_f32(acc, nb4, F, lane);              wave_sync();

    layernorm_stats(F, L.mu, L.rs, lane);    wave_sync();

    float gv[4], bv[4];
    #pragma unroll
    for (int it = 0; it < 4; ++it) { gv[it] = ng[it * 32 + lane]; bv[it] = nbeta[it * 32 + lane]; }

    for (int row = 0; row < 16; ++row) {
        const long long n = nbase + row;
        const float m = L.mu[row], r = L.rs[row];
        #pragma unroll
        for (int it = 0; it < 4; ++it) {
            const int c = it * 32 + lane;
            const float y = (F[row * 128 + c] - m) * r * gv[it] + bv[it];
            out_node[n * H + c] = x_node[n * H + c] + y;   // residual node output
        }
    }
}

// ---------------------------------------------------------------------------
// Launch: prep weights -> zero agg -> edge pass (scatter) -> node pass.
// Workspace: [ agg: 20000*128 f32 = 10.24 MB ][ wfrag: 44*4096 bf16 = 352 KB ]
// ---------------------------------------------------------------------------
extern "C" void kernel_launch(void* const* d_in, const int* in_sizes, int n_in,
                              void* d_out, int out_size, void* d_ws, size_t ws_size,
                              hipStream_t stream) {
    const float*     x_node     = (const float*)d_in[0];
    const float*     x_edge     = (const float*)d_in[1];
    const long long* edge_index = (const long long*)d_in[2];
    const float* ew1 = (const float*)d_in[3],  *eb1 = (const float*)d_in[4];
    const float* ew2 = (const float*)d_in[5],  *eb2 = (const float*)d_in[6];
    const float* ew3 = (const float*)d_in[7],  *eb3 = (const float*)d_in[8];
    const float* ew4 = (const float*)d_in[9],  *eb4 = (const float*)d_in[10];
    const float* eg  = (const float*)d_in[11], *ebeta = (const float*)d_in[12];
    const float* nw1 = (const float*)d_in[13], *nb1 = (const float*)d_in[14];
    const float* nw2 = (const float*)d_in[15], *nb2 = (const float*)d_in[16];
    const float* nw3 = (const float*)d_in[17], *nb3 = (const float*)d_in[18];
    const float* nw4 = (const float*)d_in[19], *nb4 = (const float*)d_in[20];
    const float* ng  = (const float*)d_in[21], *nbeta = (const float*)d_in[22];

    float* out_node = (float*)d_out;                       // 20000*128
    float* out_edge = out_node + (long long)N_NODES * H;   // 200000*128

    float* agg = (float*)d_ws;
    unsigned short* wfrag =
        (unsigned short*)((char*)d_ws + (size_t)N_NODES * H * sizeof(float));

    prep_weights<<<dim3(44, 8), 512, 0, stream>>>(ew1, ew2, ew3, ew4,
                                                  nw1, nw2, nw3, nw4, wfrag);

    const int aggN = N_NODES * H;
    zero_f32<<<(aggN + 255) / 256, 256, 0, stream>>>(agg, aggN);

    edge_kernel<<<N_EDGES / 32, 64, 0, stream>>>(x_node, x_edge, edge_index, wfrag,
                                                 eb1, eb2, eb3, eb4, eg, ebeta,
                                                 agg, out_edge);

    node_kernel<<<N_NODES / 32, 64, 0, stream>>>(x_node, agg, wfrag,
                                                 nb1, nb2, nb3, nb4, ng, nbeta,
                                                 out_node);
}
